// GATBlock_30202210025886
// MI455X (gfx1250) — compile-verified, matrix operand order checked by reading, not run on hardware
//
#include <hip/hip_runtime.h>
#include <hip/hip_bf16.h>
#include <math.h>

#define N_NODES 50000
#define N_EDGES 600000
#define IN_CH   128
#define OUT_CH  32
#define HEADS   4
#define HC      128           // HEADS*OUT_CH
#define NEG_SLOPE 0.2f
#define BN_EPS  1e-5f

typedef __attribute__((ext_vector_type(16))) __bf16 v16bf;
typedef __attribute__((ext_vector_type(8)))  __bf16 v8bf;
typedef __attribute__((ext_vector_type(8)))  float  v8f;

// ---- float <-> order-preserving uint mapping (for atomic max on floats) ----
__device__ __forceinline__ unsigned fmap(float f) {
  unsigned u = __float_as_uint(f);
  return (u & 0x80000000u) ? ~u : (u | 0x80000000u);
}
__device__ __forceinline__ float funmap(unsigned u) {
  return __uint_as_float((u & 0x80000000u) ? (u ^ 0x80000000u) : ~u);
}
#define EMAX_NEG_INF 0x007FFFFFu   // fmap(-inf)

// ---- f32 -> bf16 bits (round-to-nearest-even) ----
__device__ __forceinline__ unsigned short f2bf_bits(float f) {
  unsigned u = __float_as_uint(f);
  unsigned r = (u + 0x7FFFu + ((u >> 16) & 1u)) >> 16;
  return (unsigned short)r;
}

// =====================================================================
// K0: initialize accumulators
// =====================================================================
__global__ void init_kernel(unsigned* __restrict__ emax, float* __restrict__ denom,
                            float* __restrict__ out, float* __restrict__ bn) {
  int idx = blockIdx.x * blockDim.x + threadIdx.x;
  if (idx < N_NODES * HC) out[idx] = 0.0f;
  if (idx < N_NODES * HEADS) { emax[idx] = EMAX_NEG_INF; denom[idx] = 0.0f; }
  if (idx < 2 * HC) bn[idx] = 0.0f;
}

// =====================================================================
// K1a: convert x to bf16 row-major (read once, feeds GEMM as packed b128s)
// =====================================================================
__global__ void convert_x_kernel(const float* __restrict__ x,
                                 unsigned short* __restrict__ xbf) {
  int idx = blockIdx.x * blockDim.x + threadIdx.x;
  if (idx < N_NODES * IN_CH) xbf[idx] = f2bf_bits(x[idx]);
}

// =====================================================================
// K1b: pack W and res_W into the exact per-lane WMMA B-operand layout.
// Layout: Bpack[mat][ntile][kstep][lane][16 bf16]  (2*8*4*32*16 = 32768 elems)
// ISA 16-bit B layout mirror: elem i<8 -> K = ks*32 + (lane>>4)*8 + i,
//                             elem i>=8 -> K = ks*32 + 16 + (lane>>4)*8 + (i-8),
//                             N = ntile*16 + (lane&15)
// =====================================================================
__global__ void pack_B_kernel(const float* __restrict__ W,
                              const float* __restrict__ resW,
                              unsigned short* __restrict__ Bpack) {
  int idx = blockIdx.x * blockDim.x + threadIdx.x;
  if (idx >= 2 * 8 * 4 * 32 * 16) return;
  int i    = idx & 15;
  int lane = (idx >> 4) & 31;
  int ks   = (idx >> 9) & 3;
  int nt   = (idx >> 11) & 7;
  int mat  = (idx >> 14) & 1;
  int khalf = (lane >> 4) * 8;
  int k = ks * 32 + ((i < 8) ? (khalf + i) : (16 + khalf + (i - 8)));
  int n = nt * 16 + (lane & 15);
  const float* src = mat ? resW : W;
  Bpack[idx] = f2bf_bits(src[(size_t)k * HC + n]);
}

// =====================================================================
// K1c: fused dual GEMM via WMMA bf16:  xh = x@W,  res = x@res_W.
// One wave -> one 16x16 f32 tile; 16 waves/block cover all 256 output
// columns of one 16-row stripe. The shared 16x128 bf16 A-tile is staged
// in LDS via async-to-LDS copies; B comes pre-packed (two b128 per step).
// =====================================================================
__global__ __launch_bounds__(512) void gemm_wmma_kernel(
    const unsigned short* __restrict__ xbf,
    const unsigned short* __restrict__ Bpack,
    float* __restrict__ xh, float* __restrict__ res) {
  __shared__ __align__(16) unsigned short tileA[16 * IN_CH];  // 4 KB bf16 bits

  const int wave = threadIdx.x >> 5;       // 0..15
  const int lane = threadIdx.x & 31;
  const int m0 = blockIdx.x * 16;          // 3125 stripes exactly
  const int mat = (wave >> 3) & 1;         // 0 -> W/xh, 1 -> res_W/res
  const int nt  = wave & 7;
  float* __restrict__ Out = mat ? res : xh;

  // --- async stage A-tile (16 rows x 128 cols bf16 = 256 x 16B) into LDS ---
  if (threadIdx.x < 256) {
    const unsigned short* g = xbf + (size_t)m0 * IN_CH + threadIdx.x * 8;
    unsigned lds = (unsigned)(size_t)(&tileA[threadIdx.x * 8]);
    asm volatile("global_load_async_to_lds_b128 %0, %1, off"
                 :: "v"(lds), "v"(g) : "memory");
    asm volatile("s_wait_asynccnt 0x0" ::: "memory");
  }
  __syncthreads();

  const int arow  = (lane & 15) * IN_CH;
  const int khalf = (lane >> 4) * 8;
  const unsigned short* bptr =
      Bpack + (((size_t)(mat * 8 + nt) * 4) * 32 + lane) * 16;

  v8f acc = {};
  #pragma unroll
  for (int ks = 0; ks < 4; ++ks) {
    const int kk = ks * 32;
    v8bf alo = *(const v8bf*)(&tileA[arow + kk + khalf]);        // K=kk+khalf..+7
    v8bf ahi = *(const v8bf*)(&tileA[arow + kk + 16 + khalf]);   // K=kk+16+khalf..+7
    v16bf a;
    #pragma unroll
    for (int i = 0; i < 8; ++i) { a[i] = alo[i]; a[8 + i] = ahi[i]; }
    v16bf b = *(const v16bf*)(bptr + (size_t)ks * 32 * 16);
    acc = __builtin_amdgcn_wmma_f32_16x16x32_bf16(
        /*neg_a=*/false, a, /*neg_b=*/false, b,
        /*c_mod=*/(short)0, acc, /*reuse_a=*/false, /*reuse_b=*/false);
  }
  // C/D layout: VGPR r, lanes 0-15 -> M=r, lanes 16-31 -> M=8+r; N = lane&15
  const int ncol  = nt * 16 + (lane & 15);
  const int rbase = m0 + 8 * (lane >> 4);
  #pragma unroll
  for (int r = 0; r < 8; ++r)
    Out[(size_t)(rbase + r) * HC + ncol] = acc[r];
}

// =====================================================================
// K2: attention logits  a_src[n,h], a_dst[n,h]
// =====================================================================
__global__ void attn_kernel(const float* __restrict__ xh,
                            const float* __restrict__ att_src,
                            const float* __restrict__ att_dst,
                            float* __restrict__ a_src, float* __restrict__ a_dst) {
  int idx = blockIdx.x * blockDim.x + threadIdx.x;   // n*HEADS + h
  if (idx >= N_NODES * HEADS) return;
  int h = idx & (HEADS - 1);
  int n = idx >> 2;
  const float* v  = xh + (size_t)n * HC + h * OUT_CH;
  const float* as = att_src + h * OUT_CH;
  const float* ad = att_dst + h * OUT_CH;
  float s1 = 0.f, s2 = 0.f;
  #pragma unroll
  for (int c = 0; c < OUT_CH; ++c) { float xv = v[c]; s1 += xv * as[c]; s2 += xv * ad[c]; }
  a_src[idx] = s1;
  a_dst[idx] = s2;
}

// =====================================================================
// K3: per-destination segment max of leaky-relu attention scores
// =====================================================================
__global__ void edge_max_kernel(const int* __restrict__ ei,
                                const float* __restrict__ a_src,
                                const float* __restrict__ a_dst,
                                unsigned* __restrict__ emax) {
  int e = blockIdx.x * blockDim.x + threadIdx.x;
  const int ET = N_EDGES + N_NODES;
  if (e >= ET) return;
  int s, d;
  if (e < N_EDGES) { s = ei[e]; d = ei[N_EDGES + e]; } else { s = d = e - N_EDGES; }
  #pragma unroll
  for (int h = 0; h < HEADS; ++h) {
    float v = a_src[s * HEADS + h] + a_dst[d * HEADS + h];
    v = (v > 0.f) ? v : v * NEG_SLOPE;
    atomicMax(&emax[d * HEADS + h], fmap(v));
  }
}

// =====================================================================
// K4: fused exp / denom / message aggregation (unnormalized; softmax
// denominator factored out of the sum). One 128-thread block per edge.
// =====================================================================
__global__ __launch_bounds__(128) void edge_agg_kernel(
    const int* __restrict__ ei, const float* __restrict__ a_src,
    const float* __restrict__ a_dst, const unsigned* __restrict__ emax,
    const float* __restrict__ xh, float* __restrict__ denom, float* __restrict__ out) {
  int e = blockIdx.x;
  int s, d;
  if (e < N_EDGES) { s = ei[e]; d = ei[N_EDGES + e]; } else { s = d = e - N_EDGES; }
  int t = threadIdx.x;
  int h = t >> 5;
  int c = t & 31;
  float v = a_src[s * HEADS + h] + a_dst[d * HEADS + h];
  v = (v > 0.f) ? v : v * NEG_SLOPE;
  float ex = __expf(v - funmap(emax[d * HEADS + h]));
  if (c == 0) atomicAdd(&denom[d * HEADS + h], ex);
  float msg = ex * xh[(size_t)s * HC + h * OUT_CH + c];
  atomicAdd(&out[(size_t)d * HC + h * OUT_CH + c], msg);
}

// =====================================================================
// K5: normalize + bias + residual (in place into d_out), accumulate BN
// channel sums with per-block private accumulation.
// =====================================================================
__global__ __launch_bounds__(128) void finalize1_kernel(
    const float* __restrict__ denom, const float* __restrict__ bias,
    const float* __restrict__ res, float* __restrict__ out,
    float* __restrict__ bn_sum, float* __restrict__ bn_sumsq) {
  int ch = threadIdx.x;                       // 0..127
  float s = 0.f, ss = 0.f;
  for (int n = blockIdx.x; n < N_NODES; n += gridDim.x) {
    size_t o = (size_t)n * HC + ch;
    float v = out[o] / denom[n * HEADS + (ch >> 5)] + bias[ch] + res[o];
    out[o] = v;
    s += v; ss += v * v;
  }
  atomicAdd(&bn_sum[ch], s);
  atomicAdd(&bn_sumsq[ch], ss);
}

// =====================================================================
// K6: BatchNorm (batch stats) + ReLU, in place on d_out.
// =====================================================================
__global__ void finalize2_kernel(const float* __restrict__ bn_sum,
                                 const float* __restrict__ bn_sumsq,
                                 const float* __restrict__ gamma,
                                 const float* __restrict__ beta,
                                 float* __restrict__ out) {
  int idx = blockIdx.x * blockDim.x + threadIdx.x;
  if (idx >= N_NODES * HC) return;
  int ch = idx & (HC - 1);
  const float invN = 1.0f / (float)N_NODES;
  float mean = bn_sum[ch] * invN;
  float var  = bn_sumsq[ch] * invN - mean * mean;
  float v = (out[idx] - mean) * rsqrtf(var + BN_EPS) * gamma[ch] + beta[ch];
  out[idx] = (v > 0.f) ? v : 0.f;
}

// =====================================================================
extern "C" void kernel_launch(void* const* d_in, const int* in_sizes, int n_in,
                              void* d_out, int out_size, void* d_ws, size_t ws_size,
                              hipStream_t stream) {
  (void)in_sizes; (void)n_in; (void)out_size; (void)ws_size;
  const float* x       = (const float*)d_in[0];
  const int*   ei      = (const int*)  d_in[1];   // [2, E]
  const float* W       = (const float*)d_in[2];   // [128,128]
  const float* att_src = (const float*)d_in[3];   // [4,32]
  const float* att_dst = (const float*)d_in[4];
  const float* bias    = (const float*)d_in[5];   // [128]
  const float* resW    = (const float*)d_in[6];   // [128,128]
  const float* gamma   = (const float*)d_in[7];
  const float* beta    = (const float*)d_in[8];
  float* out = (float*)d_out;                     // [N, 128]

  // workspace layout
  float* ws      = (float*)d_ws;
  float* xh      = ws;                                   // N*HC f32
  float* res     = xh  + (size_t)N_NODES * HC;           // N*HC f32
  unsigned short* xbf   = (unsigned short*)(res + (size_t)N_NODES * HC); // N*IN_CH bf16
  unsigned short* Bpack = xbf + (size_t)N_NODES * IN_CH;                 // 32768 bf16
  float* a_src   = (float*)(Bpack + 2 * 8 * 4 * 32 * 16);// N*HEADS
  float* a_dst   = a_src + (size_t)N_NODES * HEADS;      // N*HEADS
  unsigned* emax = (unsigned*)(a_dst + (size_t)N_NODES * HEADS); // N*HEADS
  float* denom   = (float*)(emax + (size_t)N_NODES * HEADS);     // N*HEADS
  float* bn      = denom + (size_t)N_NODES * HEADS;      // 2*HC (sum, sumsq)

  const int ET = N_EDGES + N_NODES;

  init_kernel<<<(N_NODES * HC + 255) / 256, 256, 0, stream>>>(emax, denom, out, bn);

  convert_x_kernel<<<(N_NODES * IN_CH + 255) / 256, 256, 0, stream>>>(x, xbf);
  pack_B_kernel<<<(2 * 8 * 4 * 32 * 16 + 255) / 256, 256, 0, stream>>>(W, resW, Bpack);

  gemm_wmma_kernel<<<N_NODES / 16, 512, 0, stream>>>(xbf, Bpack, xh, res);

  attn_kernel<<<(N_NODES * HEADS + 255) / 256, 256, 0, stream>>>(
      xh, att_src, att_dst, a_src, a_dst);

  edge_max_kernel<<<(ET + 255) / 256, 256, 0, stream>>>(ei, a_src, a_dst, emax);

  edge_agg_kernel<<<ET, 128, 0, stream>>>(ei, a_src, a_dst, emax, xh, denom, out);

  finalize1_kernel<<<512, 128, 0, stream>>>(denom, bias, res, out, bn, bn + HC);

  finalize2_kernel<<<(N_NODES * HC + 255) / 256, 256, 0, stream>>>(
      bn, bn + HC, gamma, beta, out);
}
